// HyperConv2D_70411693850685
// MI455X (gfx1250) — matrix operand order repaired and checked by reading, not compiled, not run on hardware
//
#include <hip/hip_runtime.h>
#include <hip/hip_bf16.h>

// ---------- CDNA5 (gfx1250, wave32) vector types ----------
typedef __attribute__((ext_vector_type(16))) __bf16        v16bf;
typedef __attribute__((ext_vector_type(8)))  float         v8f;
typedef __attribute__((ext_vector_type(4)))  unsigned int  v4u;
typedef __attribute__((ext_vector_type(8)))  int           v8i;
typedef __attribute__((ext_vector_type(4)))  int           v4i;

// fp32 -> bf16, round-to-nearest-even
__device__ __forceinline__ unsigned short f32_to_bf16_rne(float f) {
    union { float f; unsigned int u; } c; c.f = f;
    unsigned int u = c.u;
    u += 0x7FFFu + ((u >> 16) & 1u);
    return (unsigned short)(u >> 16);
}

// =====================================================================
// Phase 1: hypernetwork -> bf16 weights, layout wgtT[kh*3+kw][cout][cin]
// =====================================================================
__global__ __launch_bounds__(256) void hyper_weights_kernel(
    const float* __restrict__ emb,   // (64,4) flat
    const float* __restrict__ w1,    // (4,64)
    const float* __restrict__ b1,    // (64)
    const float* __restrict__ w2,    // (4,144)
    const float* __restrict__ b2,    // (144)
    unsigned short* __restrict__ wgtT) // 9*128*128 bf16
{
    const int tid = blockIdx.x * 256 + threadIdx.x;
    if (tid >= 3 * 3 * 128 * 128) return;

    const int ci   = tid & 127;
    const int co   = (tid >> 7) & 127;
    const int khkw = tid >> 14;          // 0..8
    const int kh = khkw / 3, kw = khkw % 3;
    const int i = ci >> 4, ib = ci & 15;
    const int j = co >> 4, ob = co & 15;
    const int b = i * 8 + j;
    const int f = ob * 9 + kh * 3 + kw;

    float acc = b2[f];
#pragma unroll
    for (int zt = 0; zt < 4; ++zt) {
        const int q = ib * 4 + zt;
        float h = b1[q];
#pragma unroll
        for (int zz = 0; zz < 4; ++zz)
            h = fmaf(emb[b * 4 + zz], w1[zz * 64 + q], h);
        h = fmaxf(h, 0.0f);                       // relu
        acc = fmaf(h, w2[zt * 144 + f], acc);
    }
    wgtT[tid] = f32_to_bf16_rne(acc);
}

// =====================================================================
// TDM: one 1-D descriptor copying a 32 KB bf16 tap slab global -> LDS.
// D# group0: count=1, lds_addr, global_addr, type=2 ("image")
// D# group1: data_size=1 (2 bytes), tensor_dim0 = tile_dim0 = 16384,
//            tensor_dim1 = 1, no padding / multicast / iteration.
// =====================================================================
__device__ __forceinline__ void tdm_load_tap(const unsigned short* gsrc,
                                             unsigned lds_byte_off) {
    const unsigned long long ga = (unsigned long long)(uintptr_t)gsrc;
    v4u g0;
    g0.x = 1u;                                                  // count=1
    g0.y = lds_byte_off;                                        // lds_addr
    g0.z = (unsigned)(ga & 0xFFFFFFFFull);                      // global_addr lo
    g0.w = (unsigned)((ga >> 32) & 0x1FFFFFFull) | (2u << 30);  // addr[56:32]|type=2

    v8i g1;
    g1[0] = 0x00010000;  // workgroup_mask=0, data_size=1 (2B)
    g1[1] = 0x40000000;  // tensor_dim0[15:0]=0x4000 at bits [63:48]
    g1[2] = 0x00010000;  // tensor_dim0[31:16]=0, tensor_dim1=1 at [95:80]
    g1[3] = 0x40000000;  // tile_dim0=0x4000 (16384 elems) at [127:112]
    g1[4] = 0;           // tile_dim1=0 (1-D), tile_dim2=0
    g1[5] = 0x4000;      // tensor_dim0_stride (unused for 1-D)
    g1[6] = 0;
    g1[7] = 0;

    const v4i gz4 = {0, 0, 0, 0};
    const v8i gz8 = {0, 0, 0, 0, 0, 0, 0, 0};
    __builtin_amdgcn_tensor_load_to_lds(g0, g1, gz4, gz4, gz8, 0);
}

// =====================================================================
// Phase 2: implicit-GEMM conv via v_wmma_f32_16x16x32_bf16
//   One block per (n, h): M-tile = 128 w-positions (112 valid), N = 128 couts
//   Outer loop: 9 (kh,kw) taps; B tap slab DMA'd by TDM, double-buffered.
//   Inner loop: 4 chunks of 32 cin -> 8 WMMAs each.
// =====================================================================
__global__ __launch_bounds__(256) void hyperconv_wmma_kernel(
    const float* __restrict__ x,              // (16,112,112,128)
    const unsigned short* __restrict__ wgtT,  // [9][128 cout][128 cin] bf16
    const float* __restrict__ bias,           // (128)
    float* __restrict__ out)                  // (16,112,112,128)
{
    __shared__ __align__(16) unsigned short ldsA[128 * 32];       // [m][k]      8 KB
    __shared__ __align__(16) unsigned short ldsB[2][128 * 128];   // ping-pong  64 KB

    const int t    = threadIdx.x;
    const int wave = t >> 5;
    const int lane = t & 31;
    const int n    = blockIdx.x / 112;
    const int h    = blockIdx.x % 112;

    v8f acc[8] = {};   // 8 N-subtiles of 16, fp32 accumulate (64 VGPRs)

    // staging roles: each thread moves 16 elems (64B fp32 in / 32B bf16 out)
    const int sm = t >> 1;            // 0..127 : A row (m = w position)
    const int sc = (t & 1) << 4;      // 0 or 16 within the 32-wide cin chunk

    union ABfrag { uint4 q[2]; v16bf v; };

    // ---- prologue: start DMA of tap 0 weights ----
    if (wave == 0)
        tdm_load_tap(wgtT, (unsigned)(uintptr_t)&ldsB[0][0]);

#pragma unroll 1
    for (int tap = 0; tap < 9; ++tap) {
        const int buf = tap & 1;
        const int kh  = tap / 3, kw = tap % 3;
        const int h_in = h + kh - 1;

        // overlap: kick off DMA for tap+1 into the other buffer, then make
        // sure the current tap's slab has landed (TDM completes in order).
        if (wave == 0) {
            if (tap < 8) {
                tdm_load_tap(wgtT + (size_t)(tap + 1) * 128 * 128,
                             (unsigned)(uintptr_t)&ldsB[buf ^ 1][0]);
                __builtin_amdgcn_s_wait_tensorcnt(1);
            } else {
                __builtin_amdgcn_s_wait_tensorcnt(0);
            }
        }
        __syncthreads();   // publish ldsB[buf] to all waves

#pragma unroll
        for (int cc = 0; cc < 4; ++cc) {
            const int cbase = cc << 5;   // 0,32,64,96

            // ---- stage A: x[n, h_in, m+kw-1, cbase+sc .. +16] -> bf16 LDS ----
            {
                const int w_in = sm + kw - 1;
                __align__(16) unsigned short abuf[16];
                if ((unsigned)h_in < 112u && (unsigned)w_in < 112u) {
                    const float4* s4 = (const float4*)(x +
                        ((((size_t)n * 112 + h_in) * 112 + w_in) * 128 + cbase + sc));
                    __align__(16) float vals[16];
                    *(float4*)&vals[0]  = s4[0];
                    *(float4*)&vals[4]  = s4[1];
                    *(float4*)&vals[8]  = s4[2];
                    *(float4*)&vals[12] = s4[3];
#pragma unroll
                    for (int e = 0; e < 16; ++e) abuf[e] = f32_to_bf16_rne(vals[e]);
                } else {
#pragma unroll
                    for (int e = 0; e < 16; ++e) abuf[e] = 0;
                }
                uint4* d = (uint4*)&ldsA[sm * 32 + sc];
                d[0] = ((const uint4*)abuf)[0];
                d[1] = ((const uint4*)abuf)[1];
            }

            __syncthreads();

            // ---- A fragment (16x32 bf16, ISA 7.12.2 lane layout) ----
            ABfrag afr;
            {
                const int am    = (wave << 4) + (lane & 15);
                const int abase = (lane >> 4) << 3;
                afr.q[0] = *(const uint4*)&ldsA[am * 32 + abase];
                afr.q[1] = *(const uint4*)&ldsA[am * 32 + 16 + abase];
            }

            // ---- 8 WMMAs across the 128 couts ----
#pragma unroll
            for (int ns = 0; ns < 8; ++ns) {
                // B frag (32x16): lane L: col = L&15, k = (L>=16)*8 + e (e<8), 16 + (L>=16)*8 + (e-8) (e>=8)
                // ldsB is [cout][cin(128)] -> two 16B reads, 32B apart
                ABfrag bfr;
                const int bco = (ns << 4) + (lane & 15);
                const int bkb = (lane >> 4) << 3;
                const unsigned short* bp = &ldsB[buf][bco * 128 + cbase + bkb];
                bfr.q[0] = *(const uint4*)(bp);
                bfr.q[1] = *(const uint4*)(bp + 16);

                acc[ns] = __builtin_amdgcn_wmma_f32_16x16x32_bf16(
                    /*neg_a=*/false, afr.v, /*neg_b=*/false, bfr.v,
                    /*c_mod=*/(short)0, acc[ns],
                    /*reuse_a=*/false, /*reuse_b=*/false);
            }

            __syncthreads();
        }
    }

    // ---- epilogue: C/D layout -> NHWC, + bias, mask w >= 112 ----
    // lanes 0-15: N=lane,    VGPR v -> M = v
    // lanes 16-31: N=lane-16, VGPR v -> M = 8 + v
    const int col   = lane & 15;
    const int rbase = (lane >> 4) << 3;
#pragma unroll
    for (int ns = 0; ns < 8; ++ns) {
        const int   co = (ns << 4) + col;
        const float bv = bias[co];
#pragma unroll
        for (int v = 0; v < 8; ++v) {
            const int wp = (wave << 4) + rbase + v;
            if (wp < 112) {
                out[((((size_t)n * 112 + h) * 112 + wp) * 128) + co] =
                    acc[ns][v] + bv;
            }
        }
    }
}

// =====================================================================
extern "C" void kernel_launch(void* const* d_in, const int* in_sizes, int n_in,
                              void* d_out, int out_size, void* d_ws, size_t ws_size,
                              hipStream_t stream) {
    const float* x    = (const float*)d_in[0];
    const float* emb  = (const float*)d_in[1];
    const float* w1   = (const float*)d_in[2];
    const float* b1   = (const float*)d_in[3];
    const float* w2   = (const float*)d_in[4];
    const float* b2   = (const float*)d_in[5];
    const float* bias = (const float*)d_in[6];

    unsigned short* wgtT = (unsigned short*)d_ws;  // 9*128*128 bf16 = 288 KB
    float* out = (float*)d_out;

    // Phase 1: generate bf16 weights (tap-major, cout-major for TDM slabs)
    hyper_weights_kernel<<<(3 * 3 * 128 * 128 + 255) / 256, 256, 0, stream>>>(
        emb, w1, b1, w2, b2, wgtT);

    // Phase 2: WMMA implicit-GEMM conv, one block per (n, h)
    hyperconv_wmma_kernel<<<16 * 112, 256, 0, stream>>>(x, wgtT, bias, out);
}
